// BlockConv_41394894799381
// MI455X (gfx1250) — matrix operand order repaired and checked by reading, not compiled, not run on hardware
//
#include <hip/hip_runtime.h>
#include <stdint.h>

#define N_NODES 50000
#define COUT    128
#define BN_EPS  1e-5f

typedef __bf16 bf16_t;
typedef bf16_t v16bf __attribute__((ext_vector_type(16)));
typedef float  v8f   __attribute__((ext_vector_type(8)));

union BFrag { v16bf v; uint32_t u[8]; uint4 q[2]; };

// Native CDNA5 float atomic-max (segment_max is order-independent, IEEE
// maximumNumber semantics match GLOBAL_ATOMIC_MAX_NUM_F32). Forced via asm so
// it cannot fall back to a CAS loop.
__device__ __forceinline__ void atomic_max_f32(float* p, float v) {
  uint64_t a = (uint64_t)(uintptr_t)p;
  asm volatile("global_atomic_max_num_f32 %0, %1, off scope:SCOPE_DEV"
               :: "v"(a), "v"(v) : "memory");
}

__device__ __forceinline__ uint2 pack4bf(float4 v) {
  union { uint2 u2; bf16_t h[4]; } pk;
  pk.h[0] = (bf16_t)v.x; pk.h[1] = (bf16_t)v.y;
  pk.h[2] = (bf16_t)v.z; pk.h[3] = (bf16_t)v.w;
  return pk.u2;
}

// ---------------------------------------------------------------------------
// Edge-message GEMM + segment-max scatter.
// Block = 256 threads = 8 waves. Each wave owns 16 edges, computes the full
// [16 x 128] message tile with v_wmma_f32_16x16x32_bf16 and scatters it with
// global_atomic_max_num_f32 into agg[dst].
// ---------------------------------------------------------------------------
template<int CIN, int KPAD>
__global__ __launch_bounds__(256)
void msg_kernel(const float* __restrict__ feat,   // [N, CIN]
                const float* __restrict__ pos,    // [N, 3]
                const int*   __restrict__ ei,     // [2, E]
                const float* __restrict__ W,      // [CIN+3, 128]
                const float* __restrict__ bias,   // [128]
                float*       __restrict__ agg,    // [N, 128], pre-filled -inf
                int nEdges)
{
  constexpr int KROWS  = CIN + 3;
  constexpr int KSTEPS = KPAD / 32;
  constexpr int C4     = CIN / 4;

  __shared__ alignas(16) bf16_t Wsh[KPAD / 2][256];   // pair-swizzled W
  __shared__ alignas(16) bf16_t At[8][16][KPAD];      // per-wave A tiles
  __shared__ int srcS[8][16];
  __shared__ int dstS[8][16];

  const int tid   = threadIdx.x;
  const int lane  = tid & 31;
  const int wv    = tid >> 5;
  const int hi    = lane >> 4;
  const int col16 = lane & 15;

  // Stage W as bf16, pair-swizzled so (k, k+1) at a column is one aligned u32.
  for (int idx = tid; idx < KPAD * 32; idx += 256) {
    int k = idx >> 5, c4 = (idx & 31) * 4;
    float4 v = make_float4(0.f, 0.f, 0.f, 0.f);
    if (k < KROWS) v = *(const float4*)(W + k * 128 + c4);
    bf16_t* dstp = &Wsh[k >> 1][0] + (k & 1);
    dstp[(c4 + 0) * 2] = (bf16_t)v.x;
    dstp[(c4 + 1) * 2] = (bf16_t)v.y;
    dstp[(c4 + 2) * 2] = (bf16_t)v.z;
    dstp[(c4 + 3) * 2] = (bf16_t)v.w;
  }
  __syncthreads();

  const int  ebase = blockIdx.x * 128 + wv * 16;
  const bool full  = (ebase + 16) <= nEdges;

  // Stage this wave's 16 edges (wave-internal LDS ops are in-order).
  if (lane < 16) {
    int e = ebase + lane;
    int s = 0, d = 0;
    if (e < nEdges) { s = ei[e]; d = ei[nEdges + e]; }
    srcS[wv][lane] = s;
    dstS[wv][lane] = d;
  }

  // Gather node features -> A tile (float4 loads, packed bf16 stores).
  for (int idx = lane; idx < 16 * C4; idx += 32) {
    int m = idx / C4, f4 = (idx - m * C4) * 4;
    float4 v = *(const float4*)(feat + (size_t)srcS[wv][m] * CIN + f4);
    *(uint2*)&At[wv][m][f4] = pack4bf(v);
  }
  // Relative positions into columns CIN..CIN+2.
  for (int idx = lane; idx < 48; idx += 32) {
    int m = idx / 3, d3 = idx - m * 3;
    float r = pos[srcS[wv][m] * 3 + d3] - pos[dstS[wv][m] * 3 + d3];
    At[wv][m][CIN + d3] = (bf16_t)r;
  }
  // Zero-pad K to KPAD.
  for (int idx = lane; idx < 16 * (KPAD - KROWS); idx += 32) {
    int m = idx / (KPAD - KROWS);
    int f = KROWS + idx - m * (KPAD - KROWS);
    At[wv][m][f] = (bf16_t)0.0f;
  }

  // A fragments: lane<16 holds K 0..15, lane>=16 holds K 16..31 (matches B).
  BFrag afrag[KSTEPS];
#pragma unroll
  for (int s = 0; s < KSTEPS; ++s) {
    const bf16_t* base = &At[wv][col16][s * 32 + hi * 16];
    afrag[s].q[0] = *(const uint4*)(base);        // VGPR 0..3
    afrag[s].q[1] = *(const uint4*)(base + 8);    // VGPR 4..7 (K + 8)
  }

  // Per-slot destination byte offsets (dst row * 512B).
  uint32_t dstOff[8];
#pragma unroll
  for (int i = 0; i < 8; ++i) dstOff[i] = (uint32_t)dstS[wv][i + hi * 8] * (COUT * 4u);
  const uint64_t aggBase = (uint64_t)(uintptr_t)agg;

#pragma unroll
  for (int n = 0; n < 8; ++n) {
    float bval = bias[n * 16 + col16];
    v8f acc = { bval, bval, bval, bval, bval, bval, bval, bval };
#pragma unroll
    for (int s = 0; s < KSTEPS; ++s) {
      BFrag bfrag;
#pragma unroll
      for (int v = 0; v < 8; ++v) {
        int k = s * 32 + hi * 16 + 2 * v;         // B layout: lanes<16 K 0..15, else 16..31
        bfrag.u[v] = *(const uint32_t*)&Wsh[k >> 1][(n * 16 + col16) * 2];
      }
      acc = __builtin_amdgcn_wmma_f32_16x16x32_bf16(
                false, afrag[s].v, false, bfrag.v, (short)0, acc, false, false);
    }
    const uint32_t colByte = (uint32_t)(n * 16 + col16) * 4u;
    if (full) {
#pragma unroll
      for (int i = 0; i < 8; ++i)
        atomic_max_f32((float*)(uintptr_t)(aggBase + dstOff[i] + colByte), acc[i]);
    } else {
#pragma unroll
      for (int i = 0; i < 8; ++i) {
        if (ebase + i + hi * 8 < nEdges)
          atomic_max_f32((float*)(uintptr_t)(aggBase + dstOff[i] + colByte), acc[i]);
      }
    }
  }
}

// ---------------------------------------------------------------------------
// Dense skip GEMM: out = x @ Wl + bl   ([N,64] x [64,128])
// ---------------------------------------------------------------------------
__global__ __launch_bounds__(256)
void skip_gemm_kernel(const float* __restrict__ x,     // [N, 64]
                      const float* __restrict__ W,     // [64, 128]
                      const float* __restrict__ bias,  // [128]
                      float*       __restrict__ out,   // [N, 128]
                      int nRows)
{
  constexpr int KPAD = 64, KSTEPS = 2;
  __shared__ alignas(16) bf16_t Wsh[KPAD / 2][256];
  __shared__ alignas(16) bf16_t At[8][16][KPAD];

  const int tid   = threadIdx.x;
  const int lane  = tid & 31;
  const int wv    = tid >> 5;
  const int hi    = lane >> 4;
  const int col16 = lane & 15;

  for (int idx = tid; idx < KPAD * 32; idx += 256) {
    int k = idx >> 5, c4 = (idx & 31) * 4;
    float4 v = *(const float4*)(W + k * 128 + c4);
    bf16_t* dstp = &Wsh[k >> 1][0] + (k & 1);
    dstp[(c4 + 0) * 2] = (bf16_t)v.x;
    dstp[(c4 + 1) * 2] = (bf16_t)v.y;
    dstp[(c4 + 2) * 2] = (bf16_t)v.z;
    dstp[(c4 + 3) * 2] = (bf16_t)v.w;
  }
  __syncthreads();

  const int  rbase = blockIdx.x * 128 + wv * 16;
  const bool full  = (rbase + 16) <= nRows;

  for (int idx = lane; idx < 16 * 16; idx += 32) {        // 16 rows x 16 float4
    int m = idx >> 4, f4 = (idx & 15) * 4;
    int r = rbase + m; r = (r < nRows) ? r : 0;
    float4 v = *(const float4*)(x + (size_t)r * 64 + f4);
    *(uint2*)&At[wv][m][f4] = pack4bf(v);
  }

  BFrag afrag[KSTEPS];
#pragma unroll
  for (int s = 0; s < KSTEPS; ++s) {
    const bf16_t* base = &At[wv][col16][s * 32 + hi * 16];
    afrag[s].q[0] = *(const uint4*)(base);
    afrag[s].q[1] = *(const uint4*)(base + 8);
  }

#pragma unroll
  for (int n = 0; n < 8; ++n) {
    float bval = bias[n * 16 + col16];
    v8f acc = { bval, bval, bval, bval, bval, bval, bval, bval };
#pragma unroll
    for (int s = 0; s < KSTEPS; ++s) {
      BFrag bfrag;
#pragma unroll
      for (int v = 0; v < 8; ++v) {
        int k = s * 32 + hi * 16 + 2 * v;
        bfrag.u[v] = *(const uint32_t*)&Wsh[k >> 1][(n * 16 + col16) * 2];
      }
      acc = __builtin_amdgcn_wmma_f32_16x16x32_bf16(
                false, afrag[s].v, false, bfrag.v, (short)0, acc, false, false);
    }
    float* p = out + (size_t)(rbase + hi * 8) * COUT + n * 16 + col16;
    if (full) {
#pragma unroll
      for (int i = 0; i < 8; ++i) p[(size_t)i * COUT] = acc[i];  // imm-offset stores
    } else {
#pragma unroll
      for (int i = 0; i < 8; ++i)
        if (rbase + hi * 8 + i < nRows) p[(size_t)i * COUT] = acc[i];
    }
  }
}

// ---------------------------------------------------------------------------
// Elementwise / reduction helpers
// ---------------------------------------------------------------------------
__global__ void fill_neginf_kernel(uint32_t* p, int n) {
  int i = blockIdx.x * 256 + threadIdx.x;
  if (i < n) p[i] = 0xFF800000u;                 // -inf
}

// BN stats per feature; optionally cleans -inf (isolated nodes) to 0 in place.
__global__ __launch_bounds__(256)
void bn_stats_kernel(float* __restrict__ d, int nRows,
                     float* __restrict__ mean, float* __restrict__ inv,
                     int clean) {
  __shared__ float s1[256], s2[256];
  const int c = blockIdx.x, t = threadIdx.x;
  float sum = 0.f, sq = 0.f;
  for (int i = t; i < nRows; i += 256) {
    float v = d[(size_t)i * 128 + c];
    if (clean && __float_as_uint(v) == 0xFF800000u) {
      v = 0.f;
      d[(size_t)i * 128 + c] = 0.f;
    }
    sum += v; sq += v * v;
  }
  s1[t] = sum; s2[t] = sq;
  __syncthreads();
  for (int off = 128; off > 0; off >>= 1) {
    if (t < off) { s1[t] += s1[t + off]; s2[t] += s2[t + off]; }
    __syncthreads();
  }
  if (t == 0) {
    float mu  = s1[0] / (float)nRows;
    float var = s2[0] / (float)nRows - mu * mu;
    mean[c] = mu;
    inv[c]  = rsqrtf(var + BN_EPS);
  }
}

__global__ void bn_apply_kernel(const float* __restrict__ in, float* __restrict__ out,
                                const float* __restrict__ mean, const float* __restrict__ inv,
                                const float* __restrict__ g, const float* __restrict__ be,
                                const float* __restrict__ addend, int n, int doRelu) {
  int i = blockIdx.x * 256 + threadIdx.x;
  if (i >= n) return;
  int c = i & 127;
  float v = (in[i] - mean[c]) * inv[c] * g[c] + be[c];
  if (addend) v += addend[i];
  if (doRelu) v = fmaxf(v, 0.0f);
  out[i] = v;
}

// ---------------------------------------------------------------------------
extern "C" void kernel_launch(void* const* d_in, const int* in_sizes, int n_in,
                              void* d_out, int out_size, void* d_ws, size_t ws_size,
                              hipStream_t stream) {
  const float* x   = (const float*)d_in[0];
  const float* pos = (const float*)d_in[1];
  const int*   ei  = (const int*)  d_in[2];
  const float* W1  = (const float*)d_in[3];
  const float* b1  = (const float*)d_in[4];
  const float* g1  = (const float*)d_in[5];
  const float* be1 = (const float*)d_in[6];
  const float* W2  = (const float*)d_in[7];
  const float* b2  = (const float*)d_in[8];
  const float* g2  = (const float*)d_in[9];
  const float* be2 = (const float*)d_in[10];
  const float* Wl  = (const float*)d_in[11];
  const float* bl  = (const float*)d_in[12];
  const float* gl  = (const float*)d_in[13];
  const float* bel = (const float*)d_in[14];
  float* out = (float*)d_out;
  const int E = in_sizes[2] / 2;

  float* agg  = (float*)d_ws;
  float* h    = agg + (size_t)N_NODES * COUT;
  float* mean = h   + (size_t)N_NODES * COUT;
  float* inv  = mean + COUT;

  const int NE = N_NODES * COUT;
  dim3 blk(256);
  dim3 gElem((NE + 255) / 256);
  dim3 gStats(COUT);
  dim3 gSkip((N_NODES + 127) / 128);
  dim3 gMsg((E + 127) / 128);

  // skip = BN(x @ Wl + bl)   (kept in d_out until the final fuse)
  skip_gemm_kernel<<<gSkip, blk, 0, stream>>>(x, Wl, bl, out, N_NODES);
  bn_stats_kernel<<<gStats, blk, 0, stream>>>(out, N_NODES, mean, inv, 0);
  bn_apply_kernel<<<gElem, blk, 0, stream>>>(out, out, mean, inv, gl, bel, nullptr, NE, 0);

  // conv1: h = relu(BN(segmax(concat(x_j, rel) @ W1 + b1)))
  fill_neginf_kernel<<<gElem, blk, 0, stream>>>((uint32_t*)agg, NE);
  msg_kernel<64, 96><<<gMsg, blk, 0, stream>>>(x, pos, ei, W1, b1, agg, E);
  bn_stats_kernel<<<gStats, blk, 0, stream>>>(agg, N_NODES, mean, inv, 1);
  bn_apply_kernel<<<gElem, blk, 0, stream>>>(agg, h, mean, inv, g1, be1, nullptr, NE, 1);

  // conv2 + residual: out = relu(BN(segmax(concat(h_j, rel) @ W2 + b2)) + skip)
  fill_neginf_kernel<<<gElem, blk, 0, stream>>>((uint32_t*)agg, NE);
  msg_kernel<128, 160><<<gMsg, blk, 0, stream>>>(h, pos, ei, W2, b2, agg, E);
  bn_stats_kernel<<<gStats, blk, 0, stream>>>(agg, N_NODES, mean, inv, 1);
  bn_apply_kernel<<<gElem, blk, 0, stream>>>(agg, out, mean, inv, g2, be2, out, NE, 1);
}
